// SwinTransformerBlock_77171972375039
// MI455X (gfx1250) — matrix operand order repaired and checked by reading, not compiled, not run on hardware
//
#include <hip/hip_runtime.h>
#include <hip/hip_bf16.h>

typedef _Float16 f16;
typedef __attribute__((ext_vector_type(16))) _Float16 v16h;
typedef __attribute__((ext_vector_type(8)))  _Float16 v8h;
typedef __attribute__((ext_vector_type(8)))  float    v8f;

#define BIMG   32
#define HH     56
#define WWID   56
#define CC     192
#define HEADS  6
#define DH     32
#define WS7    7
#define SHIFT3 3
#define NTOK   49
#define MPAD   64
#define NWIN   64     // windows per image
#define QSCALE 0.17677669529663687f

// ---- dynamic LDS layout (bytes) ----
#define OFF_XF  0        // f32 [64][192] = 49152  (async-load staging; reused as OUT)
#define OFF_Q   49152    // f16 [64][32]  = 4096
#define OFF_K   53248    // f16 [64][32]  = 4096
#define OFF_VT  57344    // f16 [32][64]  = 4096 (V transposed: [d][token])
#define OFF_S   61440    // f32 [64][64]  = 16384
#define OFF_P   77824    // f16 [64][64]  = 8192
#define OFF_AO  86016    // f16 [64][192] = 24576
#define OFF_XH  110592   // f16 [64][192] = 24576
#define SMEM_BYTES 135168

// ---------------- WMMA fragment helpers (wave32) ----------------
// A (16-bit, 16x32 MxK): lane l: m = l&15, kh = l>>4.
//   v0..7 = A[m][kh*16 + 0..15]
__device__ __forceinline__ v16h load_a16(const f16* tile, int ld) {
  int l = threadIdx.x & 31;
  int m = l & 15, kh = l >> 4;
  const f16* p = tile + m * ld + kh * 16;
  v8h lo = *(const v8h*)p;
  v8h hi = *(const v8h*)(p + 8);
  return __builtin_shufflevector(lo, hi, 0,1,2,3,4,5,6,7,8,9,10,11,12,13,14,15);
}

// B (16-bit, 32x16 KxN) from an LDS array laid out [n][k] row-major
// (B's column n is a contiguous row): lane l: n = l&15, kh = l>>4.
__device__ __forceinline__ v16h load_b16(const f16* tile, int ld) {
  int l = threadIdx.x & 31;
  int n = l & 15, kh = l >> 4;
  return *(const v16h*)(tile + n * ld + kh * 16);
}

// B from pre-swizzled global weights: [block][lane][16] contiguous.
__device__ __forceinline__ v16h load_b_sw(const f16* wsw, int blk) {
  int l = threadIdx.x & 31;
  return *(const v16h*)(wsw + ((size_t)blk * 32 + l) * 16);
}

__device__ __forceinline__ v8f wmma16(v16h a, v16h b, v8f c) {
  return __builtin_amdgcn_wmma_f32_16x16x32_f16(false, a, false, b,
                                                (short)0, c, false, false);
}

// Async global->LDS copy of 16 bytes (GLOBAL_LOAD_ASYNC_TO_LDS_B128, ASYNCcnt).
__device__ __forceinline__ void async_copy_b128(void* lds_ptr, const void* gptr) {
  unsigned int       lds  = (unsigned int)(unsigned long long)(uintptr_t)lds_ptr;
  unsigned long long gadr = (unsigned long long)(uintptr_t)gptr;
  asm volatile("global_load_async_to_lds_b128 %0, %1, off"
               :: "v"(lds), "v"(gadr) : "memory");
}
__device__ __forceinline__ void wait_async0() {
  asm volatile("s_wait_asynccnt 0x0" ::: "memory");
}

// ---------------- weight pre-swizzle: f32 -> f16 fragment order ----------------
// qkv_w [576,192]: out = x @ qkv_w.T, so B[k][j] = qkv_w[j][k].
// Fragment-native layout: [kt(6)][gnt(36)][lane(32)][j(16)].
__global__ __launch_bounds__(256) void swin_prep_weights(
    const float* __restrict__ qkv_w, const float* __restrict__ proj_w,
    f16* __restrict__ wq, f16* __restrict__ wp) {
  int e = blockIdx.x * blockDim.x + threadIdx.x;
  const int NQ = 576 * 192;
  if (e < NQ) {
    int j = e & 15, lane = (e >> 4) & 31, rest = e >> 9;
    int gnt = rest % 36, kt = rest / 36;
    int row = gnt * 16 + (lane & 15);
    int col = kt * 32 + (lane >> 4) * 16 + j;
    wq[e] = (f16)qkv_w[row * 192 + col];
  } else if (e < NQ + 192 * 192) {
    int e2 = e - NQ;
    int j = e2 & 15, lane = (e2 >> 4) & 31, rest = e2 >> 9;
    int nt = rest % 12, kt = rest / 12;
    int row = nt * 16 + (lane & 15);
    int col = kt * 32 + (lane >> 4) * 16 + j;
    wp[e2] = (f16)proj_w[row * 192 + col];
  }
}

// ---------------- fused Swin block: one workgroup per window ----------------
__global__ __launch_bounds__(256) void swin_block_kernel(
    const float* __restrict__ xg,
    const float* __restrict__ attn_mask,   // [64][49][49]
    const float* __restrict__ qkv_b,       // [576]
    const float* __restrict__ proj_b,      // [192]
    const float* __restrict__ rel_tab,     // [169][6]
    const float* __restrict__ norm_w,
    const float* __restrict__ norm_b,
    const f16*   __restrict__ wq,          // swizzled qkv weights
    const f16*   __restrict__ wp,          // swizzled proj weights
    float* __restrict__ outg)
{
  extern __shared__ char smem[];
  float* XF  = (float*)(smem + OFF_XF);    // f32 window staging (async target)
  f16*   Qs  = (f16*)(smem + OFF_Q);
  f16*   Ks  = (f16*)(smem + OFF_K);
  f16*   VT  = (f16*)(smem + OFF_VT);
  float* S   = (float*)(smem + OFF_S);
  f16*   P   = (f16*)(smem + OFF_P);
  f16*   AO  = (f16*)(smem + OFF_AO);
  f16*   XH  = (f16*)(smem + OFF_XH);
  float* OUT = (float*)(smem + OFF_XF);    // reuses XF after conversion

  __shared__ int rowBase[MPAD];

  const int tid  = threadIdx.x;
  const int lane = tid & 31;
  const int wv   = tid >> 5;               // wave id 0..7

  const int wi   = blockIdx.x;             // window 0..2047
  const int bimg = wi >> 6;
  const int wrem = wi & 63;                // window-in-image (mask index)
  const int wy   = wrem >> 3;
  const int wx   = wrem & 7;

  // keep swizzled weights warm in GL2 (global_prefetch_b8)
  __builtin_prefetch(wq + (size_t)lane * 1024, 0, 1);
  __builtin_prefetch(wp + (size_t)lane * 1024, 0, 1);

  // ---- phase 0a: async-copy shifted window rows (f32) straight into LDS ----
  // 49 rows x 192 f32 = 49*48 x 16B transfers
  for (int i = tid; i < NTOK * 48; i += 256) {
    int t = i / 48, seg = i - t * 48;
    int ty = t / WS7, tx = t - ty * WS7;
    int oh = (wy * WS7 + ty + SHIFT3) % HH;
    int ow = (wx * WS7 + tx + SHIFT3) % WWID;
    const float* g = xg + ((bimg * HH + oh) * WWID + ow) * CC + seg * 4;
    async_copy_b128(XF + t * CC + seg * 4, g);
  }
  if (tid < MPAD) {
    int rb = 0;
    if (tid < NTOK) {
      int ty = tid / WS7, tx = tid - ty * WS7;
      int oh = (wy * WS7 + ty + SHIFT3) % HH;
      int ow = (wx * WS7 + tx + SHIFT3) % WWID;
      rb = ((bimg * HH + oh) * WWID + ow) * CC;
    }
    rowBase[tid] = rb;
  }
  wait_async0();
  __syncthreads();

  // ---- phase 0b: convert to f16 A-operand layout (rows 49..63 zero) ----
  for (int i = tid; i < MPAD * CC; i += 256) {
    int t = i / CC;
    XH[i] = (t < NTOK) ? (f16)XF[i] : (f16)0.f;
  }
  __syncthreads();

  for (int h = 0; h < HEADS; ++h) {
    // ---- phase A: per-head QKV GEMM: [64x192] x [192x96] -> Q,K,VT ----
    for (int lt = wv; lt < 24; lt += 8) {
      int mt = lt & 3, ci = lt >> 2;
      int part = ci >> 1, sub = ci & 1;         // 0=q 1=k 2=v
      int gnt  = part * 12 + h * 2 + sub;
      v8f acc = {};
      #pragma unroll
      for (int kt = 0; kt < 6; ++kt) {
        v16h a = load_a16(XH + mt * 16 * CC + kt * 32, CC);
        v16h b = load_b_sw(wq, kt * 36 + gnt);
        acc = wmma16(a, b, acc);
      }
      int n = lane & 15, mh = lane >> 4;
      int dloc = sub * 16 + n;
      float bias = qkv_b[part * CC + h * DH + dloc];
      #pragma unroll
      for (int r = 0; r < 8; ++r) {
        int m = mt * 16 + mh * 8 + r;
        float v = (m < NTOK) ? (acc[r] + bias) : 0.f;
        if (part == 0)      Qs[m * DH + dloc]   = (f16)(v * QSCALE);
        else if (part == 1) Ks[m * DH + dloc]   = (f16)v;
        else                VT[dloc * MPAD + m] = (f16)v;   // transposed store
      }
    }
    __syncthreads();

    // ---- phase B: S = (Q*scale) K^T + rel_bias + mask  (16 tiles, K=32) ----
    for (int t = wv; t < 16; t += 8) {
      int mt = t & 3, nt = t >> 2;
      v16h a = load_a16(Qs + mt * 16 * DH, DH);
      v16h b = load_b16(Ks + nt * 16 * DH, DH);
      v8f acc = {};
      acc = wmma16(a, b, acc);
      int n  = nt * 16 + (lane & 15);
      int mh = lane >> 4;
      if (n < NTOK) {
        int ny = n / WS7, nx = n - ny * WS7;
        #pragma unroll
        for (int r = 0; r < 8; ++r) {
          int m = mt * 16 + mh * 8 + r;
          if (m < NTOK) {
            int ty = m / WS7, tx = m - ty * WS7;
            int ridx = (ty - ny + WS7 - 1) * (2 * WS7 - 1) + (tx - nx + WS7 - 1);
            float bias = rel_tab[ridx * HEADS + h];
            float mk   = attn_mask[(wrem * NTOK + m) * NTOK + n];
            S[m * MPAD + n] = acc[r] + bias + mk;
          }
        }
      }
    }
    __syncthreads();

    // ---- phase C: row softmax, 4 threads per row (group shfl reductions) ----
    {
      int r = tid >> 2;        // 0..63
      int s = tid & 3;
      f16* Pr = P + r * MPAD;
      if (r < NTOK) {
        float mx = -1e30f;
        for (int c = s; c < NTOK; c += 4) mx = fmaxf(mx, S[r * MPAD + c]);
        mx = fmaxf(mx, __shfl_xor(mx, 1, 32));
        mx = fmaxf(mx, __shfl_xor(mx, 2, 32));
        float sum = 0.f;
        for (int c = s; c < NTOK; c += 4) {
          float e = __expf(S[r * MPAD + c] - mx);
          S[r * MPAD + c] = e;
          sum += e;
        }
        sum += __shfl_xor(sum, 1, 32);
        sum += __shfl_xor(sum, 2, 32);
        float inv = 1.f / sum;
        for (int c = s; c < NTOK; c += 4) Pr[c] = (f16)(S[r * MPAD + c] * inv);
        for (int c = NTOK + s; c < MPAD; c += 4) Pr[c] = (f16)0.f;
      } else {
        for (int c = s; c < MPAD; c += 4) Pr[c] = (f16)0.f;
      }
    }
    __syncthreads();

    // ---- phase D: O_h = P @ V  (8 tiles, K=64), one tile per wave ----
    {
      int mt = wv & 3, nt = wv >> 2;
      v8f acc = {};
      #pragma unroll
      for (int kt = 0; kt < 2; ++kt) {
        v16h a = load_a16(P + mt * 16 * MPAD + kt * 32, MPAD);
        v16h b = load_b16(VT + (nt * 16) * MPAD + kt * 32, MPAD);
        acc = wmma16(a, b, acc);
      }
      int n = lane & 15, mh = lane >> 4;
      int c = h * DH + nt * 16 + n;
      #pragma unroll
      for (int r = 0; r < 8; ++r) {
        int m = mt * 16 + mh * 8 + r;
        AO[m * CC + c] = (f16)acc[r];
      }
    }
    __syncthreads();
  }

  // ---- phase E: proj GEMM [64x192] x [192x192] -> OUT (reuses XF region) ----
  for (int t = wv; t < 48; t += 8) {
    int mt = t & 3, nt = t >> 2;
    v8f acc = {};
    #pragma unroll
    for (int kt = 0; kt < 6; ++kt) {
      v16h a = load_a16(AO + mt * 16 * CC + kt * 32, CC);
      v16h b = load_b_sw(wp, kt * 12 + nt);
      acc = wmma16(a, b, acc);
    }
    int n = lane & 15, mh = lane >> 4;
    int col = nt * 16 + n;
    float pb = proj_b[col];
    #pragma unroll
    for (int r = 0; r < 8; ++r) {
      int m = mt * 16 + mh * 8 + r;
      OUT[m * CC + col] = acc[r] + pb;
    }
  }
  __syncthreads();

  // ---- phase F: LayerNorm(eps=1e-6) + residual + reverse-roll scatter ----
  for (int r = wv; r < NTOK; r += 8) {
    float s0 = 0.f, s1 = 0.f;
    float vals[6];
    #pragma unroll
    for (int j = 0; j < 6; ++j) {
      int c = lane + 32 * j;
      float v = OUT[r * CC + c];
      vals[j] = v;
      s0 += v;
      s1 += v * v;
    }
    #pragma unroll
    for (int off = 16; off > 0; off >>= 1) {
      s0 += __shfl_xor(s0, off, 32);
      s1 += __shfl_xor(s1, off, 32);
    }
    float mean = s0 * (1.f / CC);
    float var  = s1 * (1.f / CC) - mean * mean;
    float rstd = rsqrtf(var + 1e-6f);
    int rb = rowBase[r];
    #pragma unroll
    for (int j = 0; j < 6; ++j) {
      int c = lane + 32 * j;
      float ln = (vals[j] - mean) * rstd * norm_w[c] + norm_b[c];
      outg[rb + c] = xg[rb + c] + ln;   // scatter == gather location (roll inverse)
    }
  }
}

extern "C" void kernel_launch(void* const* d_in, const int* in_sizes, int n_in,
                              void* d_out, int out_size, void* d_ws, size_t ws_size,
                              hipStream_t stream) {
  const float* x      = (const float*)d_in[0];
  const float* amask  = (const float*)d_in[1];
  const float* qkv_w  = (const float*)d_in[2];
  const float* qkv_b  = (const float*)d_in[3];
  const float* proj_w = (const float*)d_in[4];
  const float* proj_b = (const float*)d_in[5];
  const float* relt   = (const float*)d_in[6];
  const float* nw     = (const float*)d_in[7];
  const float* nb     = (const float*)d_in[8];
  float* out = (float*)d_out;

  f16* wq = (f16*)d_ws;                 // 576*192 halves
  f16* wp = wq + 576 * 192;             // 192*192 halves

  const int prepN = 576 * 192 + 192 * 192;
  swin_prep_weights<<<(prepN + 255) / 256, 256, 0, stream>>>(qkv_w, proj_w, wq, wp);

  swin_block_kernel<<<BIMG * NWIN, 256, SMEM_BYTES, stream>>>(
      x, amask, qkv_b, proj_b, relt, nw, nb, wq, wp, out);
}
